// Adapter_33904471835212
// MI455X (gfx1250) — matrix operand order repaired and checked by reading, not compiled, not run on hardware
//
#include <hip/hip_runtime.h>

typedef __attribute__((ext_vector_type(16))) __bf16 v16bf;
typedef __attribute__((ext_vector_type(8)))  __bf16 v8bf;
typedef __attribute__((ext_vector_type(2)))  __bf16 v2bf;
typedef __attribute__((ext_vector_type(8)))  float  v8f;
typedef unsigned short u16;
typedef unsigned int   u32;

#define C_IN 768
#define CA   384
#define T_   8
#define W_   14
#define HW_  196
#define L_   197
#define BT_  128
#define M_TOT (BT_*HW_)          // 25088
#define BM 128
#define BN 128
#define BK 32
#define KT (C_IN/BK)             // 24
#define LDP 40                   // LDS row pitch (bf16 elems): 32 + 8 pad -> 80B

// ---------- helpers (hardware bf16 conversion) ----------
static __device__ __forceinline__ u16 f2bf(float f) {
  return __builtin_bit_cast(u16, (__bf16)f);
}
static __device__ __forceinline__ float bf2f(u16 h) {
  u32 u = ((u32)h) << 16;
  return __builtin_bit_cast(float, u);
}
static __device__ __forceinline__ u32 pk2(float a, float b) {
  v2bf p;
  p.x = (__bf16)a;
  p.y = (__bf16)b;
  return __builtin_bit_cast(u32, p);
}
union AF { v16bf v; v8bf h[2]; };

// ---------- weight prep: w1T/ow1T [CA][C_IN] bf16, w2cat [C_IN n][C_IN k] bf16 ----------
__global__ void k_prep_w(const float* __restrict__ w1, const float* __restrict__ ow1,
                         const float* __restrict__ w2, const float* __restrict__ ow2,
                         u16* __restrict__ w1T, u16* __restrict__ ow1T,
                         u16* __restrict__ w2cat) {
  int idx = blockIdx.x * blockDim.x + threadIdx.x;
  const int R0 = CA * C_IN;
  if (idx < R0) {
    int n = idx / C_IN, k = idx - n * C_IN;
    w1T[idx]  = f2bf(w1[k * CA + n]);      // w1 stored [in=768][out=384]
    ow1T[idx] = f2bf(ow1[k * CA + n]);
    return;
  }
  idx -= R0;
  if (idx < C_IN * C_IN) {
    int n = idx / C_IN, k = idx - n * C_IN;
    float v = (k < CA) ? w2[k * C_IN + n] : ow2[(k - CA) * C_IN + n];
    w2cat[idx] = f2bf(v);
  }
}

// ---------- GEMM1: A built on-the-fly from x (xs, or xs - x_former if DIFF) ----------
template <int DIFF>
__global__ __launch_bounds__(256, 1) void k_gemm1(
    const float* __restrict__ x, const u16* __restrict__ WT,
    const float* __restrict__ bias, u16* __restrict__ G) {
  __shared__ u16 sA[2][BM * LDP];
  __shared__ u16 sB[2][BM * LDP];
  const int tid = threadIdx.x, lane = tid & 31, wave = tid >> 5;
  const int waveM = wave >> 1, waveN = wave & 1;
  const int m0 = blockIdx.x * BM, n0 = blockIdx.y * BN;

  // A loader: 2 passes, 4 threads/row, 8 f32 each
  const int rA = tid >> 2, cA = (tid & 3) * 8;
  long axoff[2], afoff[2]; int aldso[2];
#pragma unroll
  for (int p = 0; p < 2; ++p) {
    int r = rA + p * 64, m = m0 + r;
    int bt = m / HW_, hw = m - bt * HW_;
    int b = bt / T_, t = bt - b * T_;
    int ft = t > 0 ? t - 1 : 0;
    axoff[p] = (long)(bt * L_ + 1 + hw) * C_IN + cA;
    afoff[p] = (long)((b * T_ + ft) * L_ + 1 + hw) * C_IN + cA;
    aldso[p] = r * LDP + cA;
  }
  // B loader: 2 threads/row, 16 bf16 each (source already [N][K] bf16)
  const int rB = tid >> 1, cB = (tid & 1) * 16;
  const long boff = (long)(n0 + rB) * C_IN + cB;
  const int bldso = rB * LDP + cB;

  auto gmove = [&](int kk, int s) {   // global -> (convert) -> LDS, buffer s
    const int k = kk * BK;
#pragma unroll
    for (int p = 0; p < 2; ++p) {
      float4 a0 = *(const float4*)(x + axoff[p] + k);
      float4 a1 = *(const float4*)(x + axoff[p] + k + 4);
      if (DIFF) {
        float4 f0 = *(const float4*)(x + afoff[p] + k);
        float4 f1 = *(const float4*)(x + afoff[p] + k + 4);
        a0.x -= f0.x; a0.y -= f0.y; a0.z -= f0.z; a0.w -= f0.w;
        a1.x -= f1.x; a1.y -= f1.y; a1.z -= f1.z; a1.w -= f1.w;
      }
      uint4 apk;
      apk.x = pk2(a0.x, a0.y); apk.y = pk2(a0.z, a0.w);
      apk.z = pk2(a1.x, a1.y); apk.w = pk2(a1.z, a1.w);
      *(uint4*)&sA[s][aldso[p]] = apk;
    }
    uint4 b0 = *(const uint4*)(WT + boff + k);
    uint4 b1 = *(const uint4*)(WT + boff + k + 8);
    *(uint4*)&sB[s][bldso] = b0;
    *(uint4*)&sB[s][bldso + 8] = b1;
  };

  v8f acc[2][4];
  const v8f vzero = {0.f,0.f,0.f,0.f,0.f,0.f,0.f,0.f};
#pragma unroll
  for (int i = 0; i < 2; ++i)
#pragma unroll
    for (int j = 0; j < 4; ++j) acc[i][j] = vzero;

  const int lrow = lane & 15, lsel = lane >> 4;
  auto compute = [&](int s) {
    AF a[2], b[4];
#pragma unroll
    for (int i = 0; i < 2; ++i) {
      const u16* pa = &sA[s][(waveM * 32 + i * 16 + lrow) * LDP + lsel * 8];
      a[i].h[0] = *(const v8bf*)(pa);           // K = lsel*8 .. +7
      a[i].h[1] = *(const v8bf*)(pa + 16);      // K = 16+lsel*8 .. +7
    }
#pragma unroll
    for (int j = 0; j < 4; ++j) {
      const u16* pb = &sB[s][(waveN * 64 + j * 16 + lrow) * LDP + lsel * 8];
      b[j].h[0] = *(const v8bf*)(pb);           // K = lsel*8 .. +7
      b[j].h[1] = *(const v8bf*)(pb + 16);      // K = 16+lsel*8 .. +7
    }
#pragma unroll
    for (int i = 0; i < 2; ++i)
#pragma unroll
      for (int j = 0; j < 4; ++j)
        acc[i][j] = __builtin_amdgcn_wmma_f32_16x16x32_bf16(
            false, a[i].v, false, b[j].v, (short)0, acc[i][j], false, false);
  };

  gmove(0, 0); __syncthreads();
  int buf = 0;
  for (int kk = 0; kk < KT; ++kk) {
    if (kk + 1 < KT) gmove(kk + 1, buf ^ 1);
    compute(buf);
    if (kk + 1 < KT) { __syncthreads(); buf ^= 1; }
  }

#pragma unroll
  for (int i = 0; i < 2; ++i) {
    const int rb = m0 + waveM * 32 + i * 16 + (lsel << 3);
#pragma unroll
    for (int j = 0; j < 4; ++j) {
      const int col = n0 + waveN * 64 + j * 16 + lrow;
      const float bv = bias[col];
#pragma unroll
      for (int v = 0; v < 8; ++v)
        G[(long)(rb + v) * CA + col] = f2bf(acc[i][j][v] + bv);
    }
  }
}

// ---------- depthwise 3x3x3 conv (main), bf16 in/out, f32 accum ----------
__global__ void k_conv3(const u16* __restrict__ G1, const float* __restrict__ cw,
                        const float* __restrict__ cb, u16* __restrict__ Ccat) {
  int idx = blockIdx.x * blockDim.x + threadIdx.x;
  if (idx >= M_TOT * (CA / 2)) return;
  const int c2 = (idx % (CA / 2)) * 2;
  const int m = idx / (CA / 2);
  const int bt = m / HW_, hw = m - bt * HW_;
  const int t = bt % T_, h = hw / W_, w = hw - (hw / W_) * W_;
  float a0 = cb[c2], a1 = cb[c2 + 1];
  const float* wp0 = cw + c2 * 27;
  const float* wp1 = wp0 + 27;
#pragma unroll
  for (int dt = -1; dt <= 1; ++dt) {
    int tt = t + dt; if (tt < 0 || tt >= T_) continue;
#pragma unroll
    for (int dh = -1; dh <= 1; ++dh) {
      int hh = h + dh; if (hh < 0 || hh >= W_) continue;
#pragma unroll
      for (int dw = -1; dw <= 1; ++dw) {
        int ww = w + dw; if (ww < 0 || ww >= W_) continue;
        int mm = (bt + dt) * HW_ + hh * W_ + ww;
        u32 g = *(const u32*)(G1 + (long)mm * CA + c2);
        int wi = (dt + 1) * 9 + (dh + 1) * 3 + (dw + 1);
        a0 += wp0[wi] * bf2f((u16)g);
        a1 += wp1[wi] * bf2f((u16)(g >> 16));
      }
    }
  }
  *(u32*)(Ccat + (long)m * C_IN + c2) = pk2(a0, a1);
}

// ---------- depthwise 1x3x3 conv (offset), writes upper K-half of Ccat ----------
__global__ void k_conv2(const u16* __restrict__ G2, const float* __restrict__ ocw,
                        const float* __restrict__ ocb, u16* __restrict__ Ccat) {
  int idx = blockIdx.x * blockDim.x + threadIdx.x;
  if (idx >= M_TOT * (CA / 2)) return;
  const int c2 = (idx % (CA / 2)) * 2;
  const int m = idx / (CA / 2);
  const int bt = m / HW_, hw = m - bt * HW_;
  const int h = hw / W_, w = hw - (hw / W_) * W_;
  float a0 = ocb[c2], a1 = ocb[c2 + 1];
  const float* wp0 = ocw + c2 * 9;
  const float* wp1 = wp0 + 9;
#pragma unroll
  for (int dh = -1; dh <= 1; ++dh) {
    int hh = h + dh; if (hh < 0 || hh >= W_) continue;
#pragma unroll
    for (int dw = -1; dw <= 1; ++dw) {
      int ww = w + dw; if (ww < 0 || ww >= W_) continue;
      int mm = bt * HW_ + hh * W_ + ww;
      u32 g = *(const u32*)(G2 + (long)mm * CA + c2);
      int wi = (dh + 1) * 3 + (dw + 1);
      a0 += wp0[wi] * bf2f((u16)g);
      a1 += wp1[wi] * bf2f((u16)(g >> 16));
    }
  }
  *(u32*)(Ccat + (long)m * C_IN + CA + c2) = pk2(a0, a1);
}

// ---------- GEMM2 (fused main+offset, K=768): out = x + A@Wcat + b2 + ob2 ----------
__global__ __launch_bounds__(256, 1) void k_gemm2(
    const u16* __restrict__ Acat, const u16* __restrict__ Wcat,
    const float* __restrict__ b2, const float* __restrict__ ob2,
    const float* __restrict__ x, float* __restrict__ out) {
  __shared__ u16 sA[2][BM * LDP];
  __shared__ u16 sB[2][BM * LDP];
  const int tid = threadIdx.x, lane = tid & 31, wave = tid >> 5;
  const int waveM = wave >> 1, waveN = wave & 1;
  const int m0 = blockIdx.x * BM, n0 = blockIdx.y * BN;

  const int rA = tid >> 1, cAe = (tid & 1) * 16;
  const long aoff = (long)(m0 + rA) * C_IN + cAe;
  const int aldso = rA * LDP + cAe;
  const long boff = (long)(n0 + rA) * C_IN + cAe;

  auto gmove = [&](int kk, int s) {
    const int k = kk * BK;
    uint4 a0 = *(const uint4*)(Acat + aoff + k);
    uint4 a1 = *(const uint4*)(Acat + aoff + k + 8);
    uint4 b0 = *(const uint4*)(Wcat + boff + k);
    uint4 b1 = *(const uint4*)(Wcat + boff + k + 8);
    *(uint4*)&sA[s][aldso] = a0;
    *(uint4*)&sA[s][aldso + 8] = a1;
    *(uint4*)&sB[s][aldso] = b0;
    *(uint4*)&sB[s][aldso + 8] = b1;
  };

  v8f acc[2][4];
  const v8f vzero = {0.f,0.f,0.f,0.f,0.f,0.f,0.f,0.f};
#pragma unroll
  for (int i = 0; i < 2; ++i)
#pragma unroll
    for (int j = 0; j < 4; ++j) acc[i][j] = vzero;

  const int lrow = lane & 15, lsel = lane >> 4;
  auto compute = [&](int s) {
    AF a[2], b[4];
#pragma unroll
    for (int i = 0; i < 2; ++i) {
      const u16* pa = &sA[s][(waveM * 32 + i * 16 + lrow) * LDP + lsel * 8];
      a[i].h[0] = *(const v8bf*)(pa);
      a[i].h[1] = *(const v8bf*)(pa + 16);
    }
#pragma unroll
    for (int j = 0; j < 4; ++j) {
      const u16* pb = &sB[s][(waveN * 64 + j * 16 + lrow) * LDP + lsel * 8];
      b[j].h[0] = *(const v8bf*)(pb);
      b[j].h[1] = *(const v8bf*)(pb + 16);
    }
#pragma unroll
    for (int i = 0; i < 2; ++i)
#pragma unroll
      for (int j = 0; j < 4; ++j)
        acc[i][j] = __builtin_amdgcn_wmma_f32_16x16x32_bf16(
            false, a[i].v, false, b[j].v, (short)0, acc[i][j], false, false);
  };

  gmove(0, 0); __syncthreads();
  int buf = 0;
  for (int kk = 0; kk < KT; ++kk) {
    if (kk + 1 < KT) gmove(kk + 1, buf ^ 1);
    compute(buf);
    if (kk + 1 < KT) { __syncthreads(); buf ^= 1; }
  }

#pragma unroll
  for (int i = 0; i < 2; ++i) {
    const int rb = m0 + waveM * 32 + i * 16 + (lsel << 3);
#pragma unroll
    for (int j = 0; j < 4; ++j) {
      const int col = n0 + waveN * 64 + j * 16 + lrow;
      const float bv = b2[col] + ob2[col];
#pragma unroll
      for (int v = 0; v < 8; ++v) {
        const int row = rb + v;
        const int bt = row / HW_, hw = row - bt * HW_;
        const long o = (long)(bt * L_ + 1 + hw) * C_IN + col;
        out[o] = x[o] + acc[i][j][v] + bv;
      }
    }
  }
}

// ---------- CLS token passthrough ----------
__global__ void k_cls(const float* __restrict__ x, float* __restrict__ out) {
  int i = (blockIdx.x * blockDim.x + threadIdx.x) * 4;
  if (i >= BT_ * C_IN) return;
  const int bt = i / C_IN, c = i - bt * C_IN;
  const long o = (long)bt * L_ * C_IN + c;
  *(float4*)(out + o) = *(const float4*)(x + o);
}

extern "C" void kernel_launch(void* const* d_in, const int* in_sizes, int n_in,
                              void* d_out, int out_size, void* d_ws, size_t ws_size,
                              hipStream_t stream) {
  const float* x   = (const float*)d_in[0];
  const float* w1  = (const float*)d_in[1];
  const float* b1  = (const float*)d_in[2];
  const float* cw  = (const float*)d_in[3];
  const float* cb  = (const float*)d_in[4];
  const float* w2  = (const float*)d_in[5];
  const float* b2  = (const float*)d_in[6];
  const float* ow1 = (const float*)d_in[7];
  const float* ob1 = (const float*)d_in[8];
  const float* ocw = (const float*)d_in[9];
  const float* ocb = (const float*)d_in[10];
  const float* ow2 = (const float*)d_in[11];
  const float* ob2 = (const float*)d_in[12];
  float* out = (float*)d_out;
  char* ws = (char*)d_ws;

  // workspace layout (bytes); total = 79,429,632
  u16* w1T   = (u16*)(ws + 0);           // 384*768*2
  u16* ow1T  = (u16*)(ws + 589824);      // 384*768*2
  u16* w2cat = (u16*)(ws + 1179648);     // 768*768*2
  u16* G1    = (u16*)(ws + 2359296);     // 25088*384*2
  u16* G2    = (u16*)(ws + 21626880);    // 25088*384*2
  u16* Ccat  = (u16*)(ws + 40894464);    // 25088*768*2

  k_prep_w<<<3456, 256, 0, stream>>>(w1, ow1, w2, ow2, w1T, ow1T, w2cat);

  dim3 g1(M_TOT / BM, CA / BN);          // (196, 3)
  k_gemm1<0><<<g1, 256, 0, stream>>>(x, w1T, b1, G1);
  k_gemm1<1><<<g1, 256, 0, stream>>>(x, ow1T, ob1, G2);

  const int convN = M_TOT * (CA / 2);    // 4,816,896
  k_conv3<<<(convN + 255) / 256, 256, 0, stream>>>(G1, cw, cb, Ccat);
  k_conv2<<<(convN + 255) / 256, 256, 0, stream>>>(G2, ocw, ocb, Ccat);

  dim3 g2(M_TOT / BM, C_IN / BN);        // (196, 6)
  k_gemm2<<<g2, 256, 0, stream>>>(Ccat, w2cat, b2, ob2, x, out);

  k_cls<<<96, 256, 0, stream>>>(x, out);
}
